// GQA_36876589203951
// MI455X (gfx1250) — compile-verified, matrix operand order checked by reading, not run on hardware
//
#include <hip/hip_runtime.h>
#include <hip/hip_bf16.h>

typedef __attribute__((ext_vector_type(16))) _Float16 v16h;
typedef __attribute__((ext_vector_type(8)))  _Float16 v8h;
typedef __attribute__((ext_vector_type(8)))  float    v8f;

#define B_SZ 2
#define L_SEQ 2048
#define DIM 2048
#define NHEADS 32
#define NKV 8
#define HD 64
#define KVW (2 * NKV * HD)    // 1024
#define MTOT (B_SZ * L_SEQ)   // 4096

// ---------------------------------------------------------------------------
// CDNA5 helpers
// ---------------------------------------------------------------------------
__device__ __forceinline__ v8f wmma_f16(v16h a, v16h b, v8f c) {
  return __builtin_amdgcn_wmma_f32_16x16x32_f16(
      false, a, false, b, (short)0, c, false, false);
}

// 16x32 f16 fragment load (A layout; also serves B from a [N,K] row-major
// source). lane&15 selects row; halves follow the 0-7 / 16-23 K split, with
// lanes 16-31 shifted by 8.
__device__ __forceinline__ v16h load_frag16(const _Float16* src, int stride) {
  const int lane = threadIdx.x & 31;
  const _Float16* p = src + (size_t)(lane & 15) * stride + ((lane >> 4) << 3);
  union { v16h v; v8h h[2]; } f;
  f.h[0] = *(const v8h*)(p);
  f.h[1] = *(const v8h*)(p + 16);
  return f.v;
}

// gfx1250 async copy: 16 bytes/lane global -> LDS, tracked by ASYNCcnt.
// GVS addressing: addr = SGPR64 base + VGPR32 byte offset.
__device__ __forceinline__ void async_ld_b128(uint32_t lds_off,
                                              unsigned long long gbase,
                                              uint32_t voff) {
  asm volatile("global_load_async_to_lds_b128 %0, %1, %2"
               :: "v"(lds_off), "v"(voff), "s"(gbase)
               : "memory");
}
__device__ __forceinline__ void wait_async0() {
  asm volatile("s_wait_asynccnt 0" ::: "memory");
}

// ---------------------------------------------------------------------------
// Kernel 1: f32 -> f16 cast
// ---------------------------------------------------------------------------
__global__ void cvt_f32_f16(const float* __restrict__ in,
                            _Float16* __restrict__ out, int n) {
  int i = blockIdx.x * blockDim.x + threadIdx.x;
  if (i < n) out[i] = (_Float16)in[i];
}

// ---------------------------------------------------------------------------
// Kernel 2: C[M,N] = A[M,K] @ W[N,K]^T + bias[N], row-major operands.
// Block 128x128, 8 waves in 4(M) x 2(N); wave tile 32x64 -> 8 WMMA / K-step.
// Tiles staged with async-to-LDS, double-buffered.
// ---------------------------------------------------------------------------
template <bool HALF_OUT>
__global__ __launch_bounds__(256) void gemm_rcr(
    const _Float16* __restrict__ A, const _Float16* __restrict__ W,
    const float* __restrict__ bias, _Float16* __restrict__ outh,
    float* __restrict__ outf, int M, int N, int K) {
  __shared__ __align__(16) _Float16 As[2][128 * 32];
  __shared__ __align__(16) _Float16 Bs[2][128 * 32];

  const int t = threadIdx.x;
  const int lane = t & 31, wid = t >> 5;
  const int wm = wid >> 1, wn = wid & 1;
  const int m0 = blockIdx.y * 128, n0 = blockIdx.x * 128;

  // staging: thread t owns 32 bytes of row (t>>1), half (t&1) of the K-tile
  const int srow = t >> 1, sc0 = (t & 1) * 16;
  const uint32_t la[2] = {(uint32_t)(uintptr_t)&As[0][srow * 32 + sc0],
                          (uint32_t)(uintptr_t)&As[1][srow * 32 + sc0]};
  const uint32_t lb[2] = {(uint32_t)(uintptr_t)&Bs[0][srow * 32 + sc0],
                          (uint32_t)(uintptr_t)&Bs[1][srow * 32 + sc0]};
  const unsigned long long gA = (unsigned long long)(A + (size_t)m0 * K);
  const unsigned long long gB = (unsigned long long)(W + (size_t)n0 * K);
  const uint32_t tvoff = (uint32_t)((srow * K + sc0) * 2);

  v8f acc[2][4] = {};

  // prologue: stage tile 0
  {
    uint32_t vo = tvoff;
    async_ld_b128(la[0], gA, vo);
    async_ld_b128(la[0] + 16, gA, vo + 16);
    async_ld_b128(lb[0], gB, vo);
    async_ld_b128(lb[0] + 16, gB, vo + 16);
  }
  wait_async0();
  __syncthreads();

  const int nk = K / 32;
  for (int kt = 0; kt < nk; ++kt) {
    const int buf = kt & 1;
    if (kt + 1 < nk) {  // stage next tile while computing this one
      uint32_t vo = tvoff + (uint32_t)((kt + 1) * 32 * 2);
      async_ld_b128(la[buf ^ 1], gA, vo);
      async_ld_b128(la[buf ^ 1] + 16, gA, vo + 16);
      async_ld_b128(lb[buf ^ 1], gB, vo);
      async_ld_b128(lb[buf ^ 1] + 16, gB, vo + 16);
    }

    const _Float16* Ab = As[buf];
    const _Float16* Bb = Bs[buf];
    v16h af0 = load_frag16(&Ab[(wm * 32 + 0)  * 32], 32);
    v16h af1 = load_frag16(&Ab[(wm * 32 + 16) * 32], 32);
    v16h bf0 = load_frag16(&Bb[(wn * 64 + 0)  * 32], 32);
    v16h bf1 = load_frag16(&Bb[(wn * 64 + 16) * 32], 32);
    v16h bf2 = load_frag16(&Bb[(wn * 64 + 32) * 32], 32);
    v16h bf3 = load_frag16(&Bb[(wn * 64 + 48) * 32], 32);

    acc[0][0] = wmma_f16(af0, bf0, acc[0][0]);
    acc[0][1] = wmma_f16(af0, bf1, acc[0][1]);
    acc[0][2] = wmma_f16(af0, bf2, acc[0][2]);
    acc[0][3] = wmma_f16(af0, bf3, acc[0][3]);
    acc[1][0] = wmma_f16(af1, bf0, acc[1][0]);
    acc[1][1] = wmma_f16(af1, bf1, acc[1][1]);
    acc[1][2] = wmma_f16(af1, bf2, acc[1][2]);
    acc[1][3] = wmma_f16(af1, bf3, acc[1][3]);

    wait_async0();     // our async writes for the next buffer are done
    __syncthreads();   // every wave's writes visible -> safe to consume
  }

  const int hi = lane >> 4, col = lane & 15;
#pragma unroll
  for (int i = 0; i < 2; ++i) {
#pragma unroll
    for (int j = 0; j < 4; ++j) {
      int cn = n0 + wn * 64 + j * 16 + col;
      float bv = bias[cn];
#pragma unroll
      for (int r = 0; r < 8; ++r) {
        int rm = m0 + wm * 32 + i * 16 + r + hi * 8;
        float v = acc[i][j][r] + bv;
        if (HALF_OUT)
          outh[(size_t)rm * N + cn] = (_Float16)v;
        else
          outf[(size_t)rm * N + cn] = v;
      }
    }
  }
}

// ---------------------------------------------------------------------------
// Kernel 3: flash attention. grid = (B*H, L/128); 8 waves x 16 queries.
// K tile staged via async-to-LDS; V staged transposed ([d][key]) manually.
// ---------------------------------------------------------------------------
__global__ __launch_bounds__(256) void attn_kernel(
    const _Float16* __restrict__ Qh, const _Float16* __restrict__ KVh,
    _Float16* __restrict__ Oh, const int* __restrict__ causal_p) {
  __shared__ __align__(16) _Float16 Ks[64 * 64];
  __shared__ __align__(16) _Float16 Vs[64 * 64];  // transposed [d][key]
  __shared__ __align__(16) _Float16 Ps[8 * 16 * 32];

  const int t = threadIdx.x;
  const int lane = t & 31, wid = t >> 5;
  const int bh = blockIdx.x;
  const int b = bh >> 5, h = bh & 31, kvh = h >> 2;
  const int qblk = blockIdx.y;
  const int q0 = qblk * 128 + wid * 16;
  const int causal = *causal_p;
  const int hi = lane >> 4, col = lane & 15;
  const float scale = 0.125f;

  const size_t qbase = ((size_t)(b * L_SEQ + q0)) * DIM + h * HD;
  const v16h qf0 = load_frag16(Qh + qbase, DIM);
  const v16h qf1 = load_frag16(Qh + qbase + 32, DIM);

  v8f o0 = {}, o1 = {}, o2 = {}, o3 = {};
  float mrow[8], lrow[8];
#pragma unroll
  for (int r = 0; r < 8; ++r) { mrow[r] = -3.0e38f; lrow[r] = 0.0f; }

  _Float16* myP = Ps + wid * (16 * 32);
  const int ktEnd = causal ? (qblk * 2 + 2) : (L_SEQ / 64);

  // staging geometry: thread owns 32B of key-row (t>>2), quarter (t&3)
  const int krow = t >> 2, c0 = (t & 3) * 16;
  const uint32_t kls = (uint32_t)(uintptr_t)&Ks[krow * 64 + c0];
  const uint32_t kvoff = (uint32_t)((krow * KVW + c0) * 2);

  for (int kt = 0; kt < ktEnd; ++kt) {
    const int key0 = kt * 64;
    __syncthreads();
    {
      const unsigned long long gK = (unsigned long long)(
          KVh + ((size_t)(b * L_SEQ + key0)) * KVW + kvh * HD);
      async_ld_b128(kls, gK, kvoff);
      async_ld_b128(kls + 16, gK, kvoff + 16);
      // V: transpose into [d][key]
      const _Float16* gv = KVh + ((size_t)(b * L_SEQ + key0 + krow)) * KVW +
                           kvh * HD + NKV * HD + c0;
      v8h vlo = *(const v8h*)(gv);
      v8h vhi = *(const v8h*)(gv + 8);
#pragma unroll
      for (int j = 0; j < 8; ++j) {
        Vs[(c0 + j) * 64 + krow]     = vlo[j];
        Vs[(c0 + 8 + j) * 64 + krow] = vhi[j];
      }
    }
    wait_async0();
    __syncthreads();

#pragma unroll
    for (int sub = 0; sub < 2; ++sub) {
      v8f s0 = {}, s1 = {};
#pragma unroll
      for (int kk = 0; kk < 2; ++kk) {
        v16h qa = kk ? qf1 : qf0;
        v16h kb0 = load_frag16(&Ks[(sub * 32 + 0)  * 64 + kk * 32], 64);
        v16h kb1 = load_frag16(&Ks[(sub * 32 + 16) * 64 + kk * 32], 64);
        s0 = wmma_f16(qa, kb0, s0);
        s1 = wmma_f16(qa, kb1, s1);
      }

      const int kc0 = key0 + sub * 32 + col;
      const int kc1 = kc0 + 16;
#pragma unroll
      for (int r = 0; r < 8; ++r) {
        const int q = q0 + r + hi * 8;
        float a0 = s0[r] * scale, a1 = s1[r] * scale;
        if (causal && kc0 > q) a0 = -3.0e38f;
        if (causal && kc1 > q) a1 = -3.0e38f;
        float mnew = fmaxf(mrow[r], fmaxf(a0, a1));
#pragma unroll
        for (int msk = 1; msk < 16; msk <<= 1)
          mnew = fmaxf(mnew, __shfl_xor(mnew, msk, 32));
        const float fac = __expf(mrow[r] - mnew);
        const float p0 = __expf(a0 - mnew);
        const float p1 = __expf(a1 - mnew);
        float rs = p0 + p1;
#pragma unroll
        for (int msk = 1; msk < 16; msk <<= 1) rs += __shfl_xor(rs, msk, 32);
        lrow[r] = lrow[r] * fac + rs;
        mrow[r] = mnew;
        o0[r] = o0[r] * fac; o1[r] = o1[r] * fac;
        o2[r] = o2[r] * fac; o3[r] = o3[r] * fac;
        const int prow = r + hi * 8;
        myP[prow * 32 + col]      = (_Float16)p0;
        myP[prow * 32 + col + 16] = (_Float16)p1;
      }
      asm volatile("s_wait_dscnt 0" ::: "memory");  // wave-local P RAW

      v16h pa  = load_frag16(myP, 32);
      v16h vb0 = load_frag16(&Vs[ 0 * 64 + sub * 32], 64);
      v16h vb1 = load_frag16(&Vs[16 * 64 + sub * 32], 64);
      v16h vb2 = load_frag16(&Vs[32 * 64 + sub * 32], 64);
      v16h vb3 = load_frag16(&Vs[48 * 64 + sub * 32], 64);
      o0 = wmma_f16(pa, vb0, o0);
      o1 = wmma_f16(pa, vb1, o1);
      o2 = wmma_f16(pa, vb2, o2);
      o3 = wmma_f16(pa, vb3, o3);
    }
  }

#pragma unroll
  for (int r = 0; r < 8; ++r) {
    const float inv = 1.0f / lrow[r];
    const int q = q0 + r + hi * 8;
    const size_t ob = ((size_t)(b * L_SEQ + q)) * DIM + h * HD;
    Oh[ob + 0 * 16 + col] = (_Float16)(o0[r] * inv);
    Oh[ob + 1 * 16 + col] = (_Float16)(o1[r] * inv);
    Oh[ob + 2 * 16 + col] = (_Float16)(o2[r] * inv);
    Oh[ob + 3 * 16 + col] = (_Float16)(o3[r] * inv);
  }
}

// ---------------------------------------------------------------------------
extern "C" void kernel_launch(void* const* d_in, const int* in_sizes, int n_in,
                              void* d_out, int out_size, void* d_ws,
                              size_t ws_size, hipStream_t stream) {
  const float* x    = (const float*)d_in[0];
  const float* Wq   = (const float*)d_in[1];
  const float* bq   = (const float*)d_in[2];
  const float* Wkv  = (const float*)d_in[3];
  const float* bkv  = (const float*)d_in[4];
  const float* Wo   = (const float*)d_in[5];
  const float* bo   = (const float*)d_in[6];
  const int*   caus = (const int*)d_in[7];
  float* out = (float*)d_out;

  _Float16* xh   = (_Float16*)d_ws;
  _Float16* Wqh  = xh   + (size_t)MTOT * DIM;
  _Float16* Wkvh = Wqh  + (size_t)DIM * DIM;
  _Float16* Woh  = Wkvh + (size_t)KVW * DIM;
  _Float16* Qh   = Woh  + (size_t)DIM * DIM;
  _Float16* KVh  = Qh   + (size_t)MTOT * DIM;
  _Float16* Oh   = KVh  + (size_t)MTOT * KVW;

  const int nx  = MTOT * DIM;
  const int nwq = DIM * DIM;
  const int nkv = KVW * DIM;
  cvt_f32_f16<<<(nx  + 255) / 256, 256, 0, stream>>>(x,   xh,   nx);
  cvt_f32_f16<<<(nwq + 255) / 256, 256, 0, stream>>>(Wq,  Wqh,  nwq);
  cvt_f32_f16<<<(nkv + 255) / 256, 256, 0, stream>>>(Wkv, Wkvh, nkv);
  cvt_f32_f16<<<(nwq + 255) / 256, 256, 0, stream>>>(Wo,  Woh,  nwq);

  gemm_rcr<true><<<dim3(DIM / 128, MTOT / 128), 256, 0, stream>>>(
      xh, Wqh, bq, Qh, nullptr, MTOT, DIM, DIM);
  gemm_rcr<true><<<dim3(KVW / 128, MTOT / 128), 256, 0, stream>>>(
      xh, Wkvh, bkv, KVh, nullptr, MTOT, KVW, DIM);
  attn_kernel<<<dim3(B_SZ * NHEADS, L_SEQ / 128), 256, 0, stream>>>(
      Qh, KVh, Oh, caus);
  gemm_rcr<false><<<dim3(DIM / 128, MTOT / 128), 256, 0, stream>>>(
      Oh, Woh, bo, nullptr, out, MTOT, DIM, DIM);
}
